// Align_module_43147241455634
// MI455X (gfx1250) — matrix operand order repaired and checked by reading, not compiled, not run on hardware
//
#include <hip/hip_runtime.h>
#include <hip/hip_bf16.h>
#include <math.h>

// ---------------------------------------------------------------------------
// CDNA5 WMMA fragment types (wave32)
// ---------------------------------------------------------------------------
typedef __attribute__((ext_vector_type(16))) __bf16 v16bf;
typedef __attribute__((ext_vector_type(8)))  __bf16 v8bf;
typedef __attribute__((ext_vector_type(8)))  float  v8f;
typedef __attribute__((ext_vector_type(4)))  int    v4i;

#define CAT16(lo, hi) __builtin_shufflevector((lo), (hi), 0,1,2,3,4,5,6,7,8,9,10,11,12,13,14,15)

// Async global->LDS copy path (gfx1250 ASYNCcnt), guarded for toolchain support.
#if defined(__has_builtin)
#  if __has_builtin(__builtin_amdgcn_global_load_async_to_lds_b128) && \
      __has_builtin(__builtin_amdgcn_s_wait_asynccnt)
#    define USE_ASYNC_LDS 1
#  endif
#endif

#if defined(USE_ASYNC_LDS)
typedef __attribute__((address_space(1))) v4i gas_v4i;   // global int4
typedef __attribute__((address_space(3))) v4i las_v4i;   // LDS int4

__device__ __forceinline__ void async_copy_b128(const float* gsrc, float* ldst) {
    __builtin_amdgcn_global_load_async_to_lds_b128(
        (gas_v4i*)(gsrc),
        (las_v4i*)(ldst),
        0, 0);
}
#endif

// A-matrix (16x32 bf16) element->K mapping per ISA 7.12.2:
// lanes 0-15: e0..7 -> K=0..7, e8..15 -> K=16..23 ; lanes 16-31: +8
__device__ __forceinline__ int a_k_of(int lane, int e) {
    int kb = (lane >> 4) * 8;
    return (e < 8) ? (kb + e) : (16 + kb + (e - 8));
}
// B-matrix (32x16 bf16): same K mapping as A (e0..7 -> K=kb..kb+7, e8..15 -> K=16+kb..)
__device__ __forceinline__ int b_k_of(int lane, int e) {
    int kb = (lane >> 4) * 8;
    return (e < 8) ? (kb + e) : (16 + kb + (e - 8));
}

__device__ __forceinline__ float lrelu(float v) { return v >= 0.0f ? v : 0.2f * v; }

__device__ __forceinline__ float samp(const float* __restrict__ p, int yi, int xi,
                                      int H, int W) {
    return (yi >= 0 && yi < H && xi >= 0 && xi < W) ? p[(size_t)yi * W + xi] : 0.0f;
}

// ---------------------------------------------------------------------------
// Bilinear x2 upsample (align_corners=False) with fused scale (prev * 2.0)
// ---------------------------------------------------------------------------
__global__ __launch_bounds__(256)
void up2x2(const float* __restrict__ in, float* __restrict__ out,
           int B, int C, int Hin, int Win, float scale)
{
    int Hout = Hin * 2, Wout = Win * 2;
    size_t N = (size_t)B * C * Hout * Wout;
    size_t i = (size_t)blockIdx.x * blockDim.x + threadIdx.x;
    if (i >= N) return;
    int ox = (int)(i % Wout); size_t t = i / Wout;
    int oy = (int)(t % Hout); t /= Hout;
    int c  = (int)(t % C);    int b = (int)(t / C);
    float cy = fminf(fmaxf((oy + 0.5f) * 0.5f - 0.5f, 0.0f), (float)(Hin - 1));
    float cx = fminf(fmaxf((ox + 0.5f) * 0.5f - 0.5f, 0.0f), (float)(Win - 1));
    int y0 = (int)floorf(cy), x0 = (int)floorf(cx);
    int y1 = min(y0 + 1, Hin - 1), x1 = min(x0 + 1, Win - 1);
    float ty = cy - (float)y0, tx = cx - (float)x0;
    const float* p = in + ((size_t)b * C + c) * Hin * Win;
    float v = (p[(size_t)y0 * Win + x0] * (1.0f - ty) + p[(size_t)y1 * Win + x0] * ty) * (1.0f - tx)
            + (p[(size_t)y0 * Win + x1] * (1.0f - ty) + p[(size_t)y1 * Win + x1] * ty) * tx;
    out[i] = v * scale;
}

// ---------------------------------------------------------------------------
// 1x1 conv over virtual concat([short, long]) : GEMM M=16 px, N=32, K=64 (bf16 WMMA)
// ---------------------------------------------------------------------------
__global__ __launch_bounds__(64)
void conv1x1_wmma(const float* __restrict__ shortf, const float* __restrict__ longf,
                  const float* __restrict__ w,      // [32,64]
                  const float* __restrict__ bias,   // [32]
                  float* __restrict__ out, int B, int H, int W)
{
    const int strips_x = W >> 4;
    int s  = blockIdx.x;
    int sx = s % strips_x; int t = s / strips_x;
    int y  = t % H;        int b = t / H;
    int x0 = sx << 4;

    int lane = threadIdx.x & 31;
    int wave = threadIdx.x >> 5;          // 2 waves -> 2 N-tiles of 16
    int m    = lane & 15;
    int n    = wave * 16 + (lane & 15);

    v8f acc = {0.f, 0.f, 0.f, 0.f, 0.f, 0.f, 0.f, 0.f};
    for (int kc = 0; kc < 64; kc += 32) {
        v16bf afrag, bfrag;
        #pragma unroll
        for (int e = 0; e < 16; e++) {
            int k = kc + a_k_of(lane, e);
            const float* src = (k < 32) ? shortf : longf;
            int c = k & 31;
            afrag[e] = (__bf16)src[(((size_t)b * 32 + c) * H + y) * W + x0 + m];
        }
        #pragma unroll
        for (int e = 0; e < 16; e++) {
            int k = kc + b_k_of(lane, e);
            bfrag[e] = (__bf16)w[(size_t)n * 64 + k];
        }
        acc = __builtin_amdgcn_wmma_f32_16x16x32_bf16(false, afrag, false, bfrag,
                                                      (short)0, acc, false, false);
    }
    float bv = bias[n];
    int mBase = (lane >> 4) << 3;
    #pragma unroll
    for (int r = 0; r < 8; r++) {
        int mm = mBase + r;
        out[(((size_t)b * 32 + n) * H + y) * W + x0 + mm] = acc[r] + bv;
    }
}

// ---------------------------------------------------------------------------
// Generic 3x3 conv (pad=1) as implicit GEMM.
// Both A (activations) and B (weights) are staged in LDS in im2col [row][K]
// bf16 layout so every WMMA fragment is built from two 16B ds_load_b128 per
// operand (matches the ISA VGPR striping: A and B lane = 8+8 contiguous K).
// Input = virtual concat(in0[Cin0 ch], in1[Cin1 ch]); optional fused LeakyReLU.
// ---------------------------------------------------------------------------
__global__ __launch_bounds__(128)
void conv3x3_wmma(const float* __restrict__ in0, const float* __restrict__ in1,
                  int Cin0, int Cin1,
                  const float* __restrict__ w,      // [Cout,Cin,3,3]
                  const float* __restrict__ bias,   // [Cout]
                  float* __restrict__ out, int Cout,
                  int B, int H, int W, int applyLrelu)
{
    const int Cin = Cin0 + Cin1;
    const int K   = Cin * 9;                 // 288 or 576
    const int strips_x = W >> 4;
    int s  = blockIdx.x;
    int sx = s % strips_x; int t = s / strips_x;
    int y  = t % H;        int b = t / H;
    int x0 = sx << 4;

    __shared__ __align__(16) __bf16 ldsA[16 * 576];   // im2col activations [m][K]
    __shared__ __align__(16) __bf16 ldsB[32 * 576];   // weights [n_local][K]

    int wavesPerBlock = blockDim.x >> 5;
    int nBase0 = blockIdx.y * wavesPerBlock * 16;
    int nrows  = wavesPerBlock * 16;

    // --- im2col fill of A: 16 pixels x K ---
    for (int i = threadIdx.x; i < 16 * K; i += blockDim.x) {
        int m = i / K, k = i - m * K;
        int cin = k / 9;
        int r   = k - cin * 9;
        int gy  = y + r / 3 - 1;
        int gx  = x0 + m + (r % 3) - 1;
        float v = 0.0f;
        if (gy >= 0 && gy < H && gx >= 0 && gx < W) {
            const float* src = (cin < Cin0) ? in0 : in1;
            int c    = (cin < Cin0) ? cin  : (cin - Cin0);
            int Csrc = (cin < Cin0) ? Cin0 : Cin1;
            v = src[(((size_t)b * Csrc + c) * H + gy) * W + gx];
        }
        ldsA[i] = (__bf16)v;
    }
    // --- weight fill of B: nrows x K ---
    for (int i = threadIdx.x; i < nrows * K; i += blockDim.x) {
        int nl = i / K, k = i - nl * K;
        int n  = nBase0 + nl;
        int cin = k / 9;
        int r   = k - cin * 9;
        float wv = (n < Cout) ? w[((size_t)n * Cin + cin) * 9 + r] : 0.0f;
        ldsB[i] = (__bf16)wv;
    }
    __syncthreads();

    int lane = threadIdx.x & 31;
    int wave = threadIdx.x >> 5;
    int numNtiles = (Cout + 15) >> 4;
    int ntile = blockIdx.y * wavesPerBlock + wave;
    if (ntile >= numNtiles) return;

    int m    = lane & 15;
    int nl   = wave * 16 + (lane & 15);
    int kb_a = (lane >> 4) << 3;             // 0 or 8
    int kb_b = (lane >> 4) << 4;             // 0 or 16

    v8f acc = {0.f, 0.f, 0.f, 0.f, 0.f, 0.f, 0.f, 0.f};
    for (int kc = 0; kc < K; kc += 32) {
        const __bf16* pa = &ldsA[m * K + kc];
        const __bf16* pb = &ldsB[nl * K + kc];
        v8bf a0 = *(const v8bf*)(pa + kb_a);
        v8bf a1 = *(const v8bf*)(pa + 16 + kb_a);
        v8bf b0 = *(const v8bf*)(pb + kb_a);
        v8bf b1 = *(const v8bf*)(pb + 16 + kb_a);
        v16bf af = CAT16(a0, a1);
        v16bf bf = CAT16(b0, b1);
        acc = __builtin_amdgcn_wmma_f32_16x16x32_bf16(false, af, false, bf,
                                                      (short)0, acc, false, false);
    }
    int n = nBase0 + nl;
    if (n < Cout) {
        float bv = bias[n];
        int mBase = (lane >> 4) << 3;
        #pragma unroll
        for (int r = 0; r < 8; r++) {
            int mm = mBase + r;
            float v = acc[r] + bv;
            if (applyLrelu) v = lrelu(v);
            out[(((size_t)b * Cout + n) * H + y) * W + x0 + mm] = v;
        }
    }
}

// ---------------------------------------------------------------------------
// Fused: offset conv (32->216, WMMA into LDS; never hits HBM) + DCNv2.
// 128 threads = 4 waves. Phase A: 14 N-tiles of om over 4 waves with im2col
// A fragments (ds_load_b128 x2). dcn_w staged via async global->LDS DMA
// (ASYNCcnt) when the toolchain exposes it.
// Phase B: 16 pixels x 8 groups; sigmoid mask, bilinear gather of long_fea,
// einsum over (g,c,k) with LDS-staged dcn_w, ds_add_f32 cross-group reduce.
// ---------------------------------------------------------------------------
__global__ __launch_bounds__(128)
void dcn_fused(const float* __restrict__ off,     // [B,32,H,W]
               const float* __restrict__ xin,     // long_fea [B,32,H,W]
               const float* __restrict__ ow,      // [216,32,3,3]
               const float* __restrict__ ob,      // [216]
               const float* __restrict__ dw,      // [32,32,3,3]
               const float* __restrict__ db,      // [32]
               float* __restrict__ out, int B, int H, int W)
{
    const int K = 288;                       // 32 * 9
    const int strips_x = W >> 4;
    int s  = blockIdx.x;
    int sx = s % strips_x; int t = s / strips_x;
    int y  = t % H;        int b = t / H;
    int x0 = sx << 4;
    int tid = threadIdx.x;

    __shared__ __align__(16) __bf16 ldsA[16 * 288]; // im2col off patch
    __shared__ float  omld[224 * 16];               // om tile [n][m]
    __shared__ __align__(16) float wlds[32 * 32 * 9]; // dcn_w staged (36 KB)
    __shared__ float  accOut[16 * 32];              // [m][o]

    // --- stage dcn weights: async global->LDS DMA if available ---
#if defined(USE_ASYNC_LDS)
    for (int i = tid; i < (32 * 32 * 9) / 4; i += blockDim.x)
        async_copy_b128(dw + i * 4, wlds + i * 4);
#else
    for (int i = tid; i < 32 * 32 * 9; i += blockDim.x) wlds[i] = dw[i];
#endif

    // --- im2col stage of off patch (bf16) ---
    for (int i = tid; i < 16 * K; i += blockDim.x) {
        int m = i / K, k = i - m * K;
        int cin = k / 9;
        int r   = k - cin * 9;
        int gy  = y + r / 3 - 1;
        int gx  = x0 + m + (r % 3) - 1;
        float v = (gy >= 0 && gy < H && gx >= 0 && gx < W)
                    ? off[(((size_t)b * 32 + cin) * H + gy) * W + gx] : 0.0f;
        ldsA[i] = (__bf16)v;
    }
    for (int i = tid; i < 16 * 32; i += blockDim.x) accOut[i] = 0.0f;

#if defined(USE_ASYNC_LDS)
    __builtin_amdgcn_s_wait_asynccnt(0);
#endif
    __syncthreads();

    // --- Phase A: om = conv3x3(off; ow) + ob, 216 channels via WMMA ---
    int lane = tid & 31;
    int wave = tid >> 5;
    int m    = lane & 15;
    int kb_a = (lane >> 4) << 3;
    for (int ntile = wave; ntile < 14; ntile += 4) {
        int n = (ntile << 4) + (lane & 15);
        v8f acc = {0.f, 0.f, 0.f, 0.f, 0.f, 0.f, 0.f, 0.f};
        for (int kc = 0; kc < K; kc += 32) {
            const __bf16* pa = &ldsA[m * K + kc];
            v8bf a0 = *(const v8bf*)(pa + kb_a);
            v8bf a1 = *(const v8bf*)(pa + 16 + kb_a);
            v16bf af = CAT16(a0, a1);
            v16bf bf;
            #pragma unroll
            for (int e = 0; e < 16; e++) {
                int k   = kc + b_k_of(lane, e);
                int cin = k / 9;
                int r   = k - cin * 9;
                float wv = (n < 216) ? ow[((size_t)n * 32 + cin) * 9 + r] : 0.0f;
                bf[e] = (__bf16)wv;
            }
            acc = __builtin_amdgcn_wmma_f32_16x16x32_bf16(false, af, false, bf,
                                                          (short)0, acc, false, false);
        }
        if (n < 216) {
            float bv = ob[n];
            int mBase = (lane >> 4) << 3;
            #pragma unroll
            for (int r = 0; r < 8; r++)
                omld[n * 16 + mBase + r] = acc[r] + bv;
        }
    }
    __syncthreads();

    // --- Phase B: modulated deformable conv ---
    int mm = tid & 15;        // pixel in strip
    int g  = tid >> 4;        // deform group 0..7
    int x  = x0 + mm;

    float partial[32];
    #pragma unroll
    for (int o = 0; o < 32; o++) partial[o] = 0.0f;

    for (int k = 0; k < 9; k++) {
        int ch = g * 9 + k;
        float dy  = omld[ch * 16 + mm];
        float dx  = omld[(72 + ch) * 16 + mm];
        float msk = 1.0f / (1.0f + expf(-omld[(144 + ch) * 16 + mm]));
        float py = dy + (float)y + (float)(k / 3 - 1);
        float px = dx + (float)x + (float)(k % 3 - 1);
        float fy = floorf(py), fx = floorf(px);
        int y0i = (int)fy, x0i = (int)fx;
        float ty = py - fy, tx = px - fx;
        float w00 = (1.0f - ty) * (1.0f - tx), w01 = (1.0f - ty) * tx;
        float w10 = ty * (1.0f - tx),          w11 = ty * tx;
        #pragma unroll
        for (int c = 0; c < 4; c++) {
            const float* p = xin + ((size_t)b * 32 + g * 4 + c) * H * W;
            float v00 = samp(p, y0i,     x0i,     H, W);
            float v01 = samp(p, y0i,     x0i + 1, H, W);
            float v10 = samp(p, y0i + 1, x0i,     H, W);
            float v11 = samp(p, y0i + 1, x0i + 1, H, W);
            float val = (w00 * v00 + w01 * v01 + w10 * v10 + w11 * v11) * msk;
            #pragma unroll
            for (int o = 0; o < 32; o++)
                partial[o] += wlds[((o * 32) + (g * 4 + c)) * 9 + k] * val;
        }
    }
    #pragma unroll
    for (int o = 0; o < 32; o++)
        atomicAdd(&accOut[mm * 32 + o], partial[o]);   // ds_add_f32
    __syncthreads();

    // --- store: bias + LeakyReLU ---
    for (int i = tid; i < 16 * 32; i += blockDim.x) {
        int m2 = i >> 5, o = i & 31;
        float v = lrelu(accOut[i] + db[o]);
        out[(((size_t)b * 32 + o) * H + y) * W + x0 + m2] = v;
    }
}

// ---------------------------------------------------------------------------
// Host-side launch
// ---------------------------------------------------------------------------
extern "C" void kernel_launch(void* const* d_in, const int* in_sizes, int n_in,
                              void* d_out, int out_size, void* d_ws, size_t ws_size,
                              hipStream_t stream)
{
    const float* short_fea = (const float*)d_in[0];
    const float* long_fea  = (const float*)d_in[1];
    const float* prev_off  = (const float*)d_in[2];
    const float* conv1_w   = (const float*)d_in[3];
    const float* conv1_b   = (const float*)d_in[4];
    const float* oc1_w     = (const float*)d_in[5];
    const float* oc1_b     = (const float*)d_in[6];
    const float* oc2_w     = (const float*)d_in[7];
    const float* oc2_b     = (const float*)d_in[8];
    const float* oc3_w     = (const float*)d_in[9];
    const float* oc3_b     = (const float*)d_in[10];
    const float* dcn_off_w = (const float*)d_in[11];
    const float* dcn_off_b = (const float*)d_in[12];
    const float* dcn_w     = (const float*)d_in[13];
    const float* dcn_b     = (const float*)d_in[14];

    const int B = 2, C = 32, H = 256, W = 256;
    const size_t plane = (size_t)B * C * H * W;   // 4,194,304 floats

    float* t0   = (float*)d_ws;      // conv1x1 out
    float* t1   = t0  + plane;       // oc1 out
    float* tup  = t1  + plane;       // upsampled prev*2
    float* t3   = tup + plane;       // oc2 out
    float* outp = (float*)d_out;     // output[0]: aligned features
    float* offp = outp + plane;      // output[1]: final offset features

    const int strips = B * H * (W / 16);  // 8192 strips of 16 px

    // 1) bilinear x2 upsample of prev_offset, fused *2.0
    {
        size_t N = plane;
        int thr = 256;
        int blk = (int)((N + thr - 1) / thr);
        up2x2<<<blk, thr, 0, stream>>>(prev_off, tup, B, C, H / 2, W / 2, 2.0f);
    }
    // 2) off = conv1x1(concat(short,long))       (no activation)
    conv1x1_wmma<<<strips, 64, 0, stream>>>(short_fea, long_fea, conv1_w, conv1_b,
                                            t0, B, H, W);
    // 3) off = lrelu(conv3x3_oc1(off))
    conv3x3_wmma<<<dim3(strips, 1), 64, 0, stream>>>(t0, nullptr, 32, 0,
                                                     oc1_w, oc1_b, t1, 32, B, H, W, 1);
    // 4) off = lrelu(conv3x3_oc2(concat(off, up2(prev)*2)))
    conv3x3_wmma<<<dim3(strips, 1), 64, 0, stream>>>(t1, tup, 32, 32,
                                                     oc2_w, oc2_b, t3, 32, B, H, W, 1);
    // 5) off = lrelu(conv3x3_oc3(off))  -> second output
    conv3x3_wmma<<<dim3(strips, 1), 64, 0, stream>>>(t3, nullptr, 32, 0,
                                                     oc3_w, oc3_b, offp, 32, B, H, W, 1);
    // 6) out = lrelu(DCNv2(long_fea, off))  (offset conv fused, om never in HBM)
    dcn_fused<<<strips, 128, 0, stream>>>(offp, long_fea, dcn_off_w, dcn_off_b,
                                          dcn_w, dcn_b, outp, B, H, W);
}
